// PerformerAttention_51728586113703
// MI455X (gfx1250) — matrix operand order repaired and checked by reading, not compiled, not run on hardware
//
#include <hip/hip_runtime.h>
#include <hip/hip_bf16.h>
#include <math.h>

// Performer (FAVOR+) causal linear attention, chunked formulation for CDNA5.
// B=8, T=8192, D=64, M=256, chunk C=128, NC=64 chunks per batch.
// The chunk GEMMs run on v_wmma_f32_16x16x32_f16 (f16 in, f32 accum).
// V^T chunk is staged into LDS with GLOBAL_LOAD_ASYNC_TO_LDS_B128 so the
// copy overlaps the QK^T WMMA phase (ASYNCcnt pipeline).

#define B_DIM 8
#define T_DIM 8192
#define D_HEAD 64
#define M_FEAT 256
#define CHUNK 128
#define NCHUNK (T_DIM / CHUNK)
#define EPSV 1e-6f

typedef __attribute__((ext_vector_type(16))) _Float16 v16h;
typedef __attribute__((ext_vector_type(8)))  _Float16 v8h;
typedef __attribute__((ext_vector_type(8)))  float    v8f;

__device__ inline v8f wmma_f16(v16h a, v16h b, v8f c) {
  // (neg_a, A, neg_b, B, c_mod, C, reuse_a, reuse_b)
  return __builtin_amdgcn_wmma_f32_16x16x32_f16(false, a, false, b, (short)0, c,
                                                false, false);
}

// A-fragment (16x32 f16, M x K): lane holds row M = lane%16.
// Halves: K in [base, base+8) then [base+16, base+24), base = (lane>>4)*8.
__device__ inline v16h load_a_frag(const _Float16* row, int kk, int lane) {
  int base = kk + ((lane >> 4) << 3);
  union { v16h v; v8h h[2]; } u;
  u.h[0] = *(const v8h*)(row + base);
  u.h[1] = *(const v8h*)(row + base + 16);
  return u.v;
}

// B-fragment (32x16 f16, K x N): lane holds column N = lane%16.
// Halves: K in [base, base+8) then [base+16, base+24), base = (lane>>4)*8.
__device__ inline v16h load_b_frag(const _Float16* col, int kk, int lane) {
  int base = kk + ((lane >> 4) << 3);
  union { v16h v; v8h h[2]; } u;
  u.h[0] = *(const v8h*)(col + base);
  u.h[1] = *(const v8h*)(col + base + 16);
  return u.v;
}

// -------------------------------------------------------------------------
// Kernel A: phi(x) = exp(x*scale @ omega^T - ||x*scale||^2/2 - rowmax)/16
// grid.x = (B*T)/8 blocks of 256 threads (8 waves, one row per wave),
// grid.y = 0 -> q path, 1 -> k path (k path also emits phi_k^T and V^T).
// -------------------------------------------------------------------------
__global__ __launch_bounds__(256) void phi_kernel(
    const float* __restrict__ q, const float* __restrict__ k,
    const float* __restrict__ v, const float* __restrict__ omega,
    _Float16* __restrict__ phiq, _Float16* __restrict__ phik,
    _Float16* __restrict__ phikT, _Float16* __restrict__ vT) {
  int w = threadIdx.x >> 5, lane = threadIdx.x & 31;
  int r = blockIdx.x * 8 + w;  // global row in [0, B*T)
  bool is_k = (blockIdx.y == 1);
  const float* x_in = is_k ? k : q;

  __shared__ float xrow[8][64];
  float x0 = x_in[(size_t)r * 64 + lane] * 0.125f;        // scale = 1/sqrt(64)
  float x1 = x_in[(size_t)r * 64 + 32 + lane] * 0.125f;
  xrow[w][lane] = x0;
  xrow[w][lane + 32] = x1;
  float ss = x0 * x0 + x1 * x1;
#pragma unroll
  for (int off = 16; off >= 1; off >>= 1) ss += __shfl_xor(ss, off, 32);
  __syncthreads();

  float logp[8];
  float mx = -1e30f;
#pragma unroll
  for (int i = 0; i < 8; ++i) {
    int f = lane + 32 * i;
    const float* om = omega + (size_t)f * 64;
    float dot = 0.f;
#pragma unroll
    for (int d = 0; d < 64; ++d) dot += xrow[w][d] * om[d];
    logp[i] = dot - 0.5f * ss;
    mx = fmaxf(mx, logp[i]);
  }
#pragma unroll
  for (int off = 16; off >= 1; off >>= 1) mx = fmaxf(mx, __shfl_xor(mx, off, 32));

  int bb = r / T_DIM, t = r % T_DIM;
#pragma unroll
  for (int i = 0; i < 8; ++i) {
    int f = lane + 32 * i;
    _Float16 ph = (_Float16)(expf(logp[i] - mx) * 0.0625f);  // /sqrt(256)
    if (is_k) {
      phik[(size_t)r * M_FEAT + f] = ph;
      phikT[((size_t)bb * M_FEAT + f) * T_DIM + t] = ph;
    } else {
      phiq[(size_t)r * M_FEAT + f] = ph;
    }
  }
  if (is_k) {
    vT[((size_t)bb * D_HEAD + lane) * T_DIM + t] =
        (_Float16)v[(size_t)r * 64 + lane];
    vT[((size_t)bb * D_HEAD + lane + 32) * T_DIM + t] =
        (_Float16)v[(size_t)r * 64 + 32 + lane];
  }
}

// -------------------------------------------------------------------------
// Kernel B: per-chunk G = phi_k^T @ V (stored d-major as G^T[d][m]) and
// zsum[m] = sum_t phi_k[t][m].  grid = (NCHUNK, B), 256 threads = 8 waves.
// -------------------------------------------------------------------------
__global__ __launch_bounds__(256) void chunk_sum_kernel(
    const _Float16* __restrict__ phikT, const _Float16* __restrict__ vT,
    float* __restrict__ GT, float* __restrict__ zsum) {
  int c = blockIdx.x, b = blockIdx.y;
  int tid = threadIdx.x, lane = tid & 31, w = tid >> 5;
  const _Float16* kTb = phikT + (size_t)b * M_FEAT * T_DIM;
  const _Float16* vTb = vT + (size_t)b * D_HEAD * T_DIM;
  float* GTc = GT + ((size_t)(b * NCHUNK + c)) * (D_HEAD * M_FEAT);

#pragma unroll
  for (int p = 0; p < 2; ++p) {
    int mi = w * 2 + p;  // 16 tiles of 16 rows cover M=256
    const _Float16* arow =
        kTb + (size_t)(mi * 16 + (lane & 15)) * T_DIM + (size_t)c * CHUNK;
    for (int dj = 0; dj < 4; ++dj) {
      v8f acc = {};
      const _Float16* vcol =
          vTb + (size_t)(dj * 16 + (lane & 15)) * T_DIM + (size_t)c * CHUNK;
#pragma unroll
      for (int kk = 0; kk < CHUNK; kk += 32)
        acc = wmma_f16(load_a_frag(arow, kk, lane),
                       load_b_frag(vcol, kk, lane), acc);
      int n = lane & 15, mbase = (lane >> 4) << 3;
#pragma unroll
      for (int j = 0; j < 8; ++j) {
        int m = mi * 16 + j + mbase;
        int d = dj * 16 + n;
        GTc[(size_t)d * M_FEAT + m] = acc[j];  // d-major (S^T layout)
      }
    }
  }
  // zsum over the chunk (one feature per thread)
  {
    const _Float16* kr = kTb + (size_t)tid * T_DIM + (size_t)c * CHUNK;
    float s = 0.f;
    for (int i = 0; i < CHUNK; ++i) s += (float)kr[i];
    zsum[((size_t)(b * NCHUNK + c)) * M_FEAT + tid] = s;
  }
}

// -------------------------------------------------------------------------
// Kernel C: exclusive prefix over chunks; state emitted directly as f16
// (WMMA input precision) so the hot consumer loads plain f16 B-fragments.
// Accumulation stays f32 in registers.  grid = B blocks.
// -------------------------------------------------------------------------
__global__ __launch_bounds__(256) void prefix_kernel(
    const float* __restrict__ GT, const float* __restrict__ zsum,
    _Float16* __restrict__ SpT16, float* __restrict__ zpref) {
  int b = blockIdx.x, tid = threadIdx.x;
  const int NS = D_HEAD * M_FEAT;  // 16384 per chunk state
  for (int e = tid; e < NS; e += 256) {
    float acc = 0.f;
    for (int c = 0; c < NCHUNK; ++c) {
      size_t idx = ((size_t)(b * NCHUNK + c)) * NS + e;
      SpT16[idx] = (_Float16)acc;
      acc += GT[idx];
    }
  }
  {
    float acc = 0.f;
    for (int c = 0; c < NCHUNK; ++c) {
      size_t idx = ((size_t)(b * NCHUNK + c)) * M_FEAT + tid;
      zpref[idx] = acc;
      acc += zsum[idx];
    }
  }
}

// -------------------------------------------------------------------------
// Kernel D: per-chunk output.
//   async-copy V^T chunk -> LDS (overlaps QK^T phase, ASYNCcnt tracked)
//   A = phi_q @ phi_k^T (masked, inclusive causal) -> LDS (f16)
//   out = tril(A)@V + phi_q@S_pref ; denom = rowsum(tril(A)) + phi_q.z_pref
// grid = (NCHUNK, B), 256 threads = 8 waves; wave w owns row-tile ti = w.
// -------------------------------------------------------------------------
__global__ __launch_bounds__(256) void chunk_out_kernel(
    const _Float16* __restrict__ phiq, const _Float16* __restrict__ phik,
    const _Float16* __restrict__ vT, const _Float16* __restrict__ SpT16,
    const float* __restrict__ zpref, float* __restrict__ out) {
  int c = blockIdx.x, b = blockIdx.y;
  int tid = threadIdx.x, lane = tid & 31, w = tid >> 5;

  __shared__ __align__(16) _Float16 As[CHUNK][CHUNK + 8];   // 34,816 B
  __shared__ __align__(16) _Float16 vsh[D_HEAD][CHUNK + 8]; // 17,408 B
  __shared__ float zsh[M_FEAT];
  __shared__ float denom[CHUNK];

  // ---- kick off async V^T chunk copy: 64 rows x 256 B, 16 B per lane ----
  {
    const _Float16* vsrc =
        vT + (size_t)b * D_HEAD * T_DIM + (size_t)c * CHUNK;
    unsigned lds_base = (unsigned)(uintptr_t)(&vsh[0][0]);
#pragma unroll
    for (int i = 0; i < 4; ++i) {
      int q16 = tid + 256 * i;   // 1024 chunks of 16 B total
      int row = q16 >> 4;        // d in [0,64)
      int col = q16 & 15;        // 16-B chunk within row
      const _Float16* src = vsrc + (size_t)row * T_DIM + col * 8;
      unsigned dst = lds_base + (unsigned)(row * (CHUNK + 8) + col * 8) * 2u;
      asm volatile("global_load_async_to_lds_b128 %0, %1, off"
                   :: "v"(dst), "v"(src) : "memory");
    }
  }

  zsh[tid] = zpref[((size_t)(b * NCHUNK + c)) * M_FEAT + tid];

  const _Float16* phiq_base = phiq + ((size_t)(b * T_DIM + c * CHUNK)) * M_FEAT;
  const _Float16* phik_base = phik + ((size_t)(b * T_DIM + c * CHUNK)) * M_FEAT;

  // ---- step 1: masked A = phi_q @ phi_k^T into LDS ----
  int ti = w;
  int arow = ti * 16 + (lane & 15);
  const _Float16* arowp = phiq_base + (size_t)arow * M_FEAT;
  for (int tj = 0; tj <= ti; ++tj) {
    v8f acc = {};
    const _Float16* bcol = phik_base + (size_t)(tj * 16 + (lane & 15)) * M_FEAT;
#pragma unroll
    for (int kk = 0; kk < M_FEAT; kk += 32)
      acc = wmma_f16(load_a_frag(arowp, kk, lane),
                     load_b_frag(bcol, kk, lane), acc);
    int n = lane & 15, mbase = (lane >> 4) << 3;
#pragma unroll
    for (int j = 0; j < 8; ++j) {
      int m = j + mbase;
      float val = acc[j];
      if (tj == ti && n > m) val = 0.f;  // inclusive causal mask
      As[ti * 16 + m][tj * 16 + n] = (_Float16)val;
    }
  }
  for (int tj = ti + 1; tj < 8; ++tj) {  // strictly-future tiles are zero
    int n = lane & 15, mbase = (lane >> 4) << 3;
#pragma unroll
    for (int j = 0; j < 8; ++j) As[ti * 16 + j + mbase][tj * 16 + n] = (_Float16)0.f;
  }

  // Async V^T copy must be complete (and visible to all waves) before the
  // intra GEMM reads vsh; As stores ride the normal DScnt + barrier.
  asm volatile("s_wait_asynccnt 0" ::: "memory");
  __syncthreads();

  // ---- denominator: rowsum(tril(A)) + phi_q . z_pref + eps ----
  if (tid < CHUNK) {
    float rs = 0.f;
    for (int i = 0; i < CHUNK; ++i) rs += (float)As[tid][i];
    const _Float16* qr = phiq_base + (size_t)tid * M_FEAT;
    float dz = 0.f;
    for (int m = 0; m < M_FEAT; ++m) dz += (float)qr[m] * zsh[m];
    denom[tid] = rs + dz + EPSV;
  }
  __syncthreads();

  // ---- step 2+3: out = tril(A)@V + phi_q@S_pref ----
  const _Float16* SpTb = SpT16 + ((size_t)(b * NCHUNK + c)) * (D_HEAD * M_FEAT);
  const _Float16* alds = &As[ti * 16 + (lane & 15)][0];  // LDS A-frag rows
  for (int dj = 0; dj < 4; ++dj) {
    v8f acc = {};
    int d0 = dj * 16 + (lane & 15);
    const _Float16* vcol = &vsh[d0][0];           // LDS B-frag columns
#pragma unroll
    for (int kk = 0; kk < CHUNK; kk += 32)        // intra, K = chunk time
      acc = wmma_f16(load_a_frag(alds, kk, lane),
                     load_b_frag(vcol, kk, lane), acc);
    const _Float16* scol = SpTb + (size_t)d0 * M_FEAT;
#pragma unroll
    for (int kk = 0; kk < M_FEAT; kk += 32)       // inter, K = feature
      acc = wmma_f16(load_a_frag(arowp, kk, lane),
                     load_b_frag(scol, kk, lane), acc);
    int n = lane & 15, mbase = (lane >> 4) << 3;
#pragma unroll
    for (int j = 0; j < 8; ++j) {
      int tl = ti * 16 + j + mbase;
      out[((size_t)(b * T_DIM + c * CHUNK + tl)) * D_HEAD + dj * 16 + n] =
          acc[j] / denom[tl];
    }
  }
}

// -------------------------------------------------------------------------
// Host launcher. Workspace layout (bytes):
//   phiq  f16 @ 0          (33,554,432)
//   phik  f16 @ 33554432   (33,554,432)
//   phikT f16 @ 67108864   (33,554,432)
//   vT    f16 @ 100663296  ( 8,388,608)
//   GT    f32 @ 109051904  (33,554,432)
//   SpT16 f16 @ 142606336  (16,777,216)
//   zsum  f32 @ 159383552  (   524,288)
//   zpref f32 @ 159907840  (   524,288)   total ~160 MB
// -------------------------------------------------------------------------
extern "C" void kernel_launch(void* const* d_in, const int* in_sizes, int n_in,
                              void* d_out, int out_size, void* d_ws,
                              size_t ws_size, hipStream_t stream) {
  (void)in_sizes; (void)n_in; (void)out_size; (void)ws_size;
  const float* q = (const float*)d_in[0];
  const float* k = (const float*)d_in[1];
  const float* v = (const float*)d_in[2];
  const float* omega = (const float*)d_in[3];
  float* out = (float*)d_out;

  char* ws = (char*)d_ws;
  _Float16* phiq  = (_Float16*)(ws + 0);
  _Float16* phik  = (_Float16*)(ws + 33554432);
  _Float16* phikT = (_Float16*)(ws + 67108864);
  _Float16* vT    = (_Float16*)(ws + 100663296);
  float*    GT    = (float*)(ws + 109051904);
  _Float16* SpT16 = (_Float16*)(ws + 142606336);
  float*    zsum  = (float*)(ws + 159383552);
  float*    zpref = (float*)(ws + 159907840);

  dim3 gA((B_DIM * T_DIM) / 8, 2);
  phi_kernel<<<gA, 256, 0, stream>>>(q, k, v, omega, phiq, phik, phikT, vT);

  dim3 gBC(NCHUNK, B_DIM);
  chunk_sum_kernel<<<gBC, 256, 0, stream>>>(phikT, vT, GT, zsum);
  prefix_kernel<<<B_DIM, 256, 0, stream>>>(GT, zsum, SpT16, zpref);
  chunk_out_kernel<<<gBC, 256, 0, stream>>>(phiq, phik, vT, SpT16, zpref, out);
}
